// DCNv2_14078902797043
// MI455X (gfx1250) — matrix operand order repaired and checked by reading, not compiled, not run on hardware
//
#include <hip/hip_runtime.h>
#include <hip/hip_bf16.h>

#define B_   4
#define C_   256
#define H_   64
#define W_   64
#define HW_  4096
#define OC_  256
#define KDIM 2304   // 9 * 256, k = tap*256 + c

typedef __attribute__((ext_vector_type(16))) __bf16 v16bf;
typedef __attribute__((ext_vector_type(8)))  float  v8f;

static __device__ __forceinline__ unsigned short f2bfu(float f) {
  unsigned u = __float_as_uint(f);
  return (unsigned short)((u + 0x7FFFu + ((u >> 16) & 1u)) >> 16);
}
static __device__ __forceinline__ __bf16 bfbits(unsigned short s) {
  union { unsigned short u; __bf16 b; } x; x.u = s; return x.b;
}

// ---- weight reorder/convert kernels --------------------------------------
__global__ void reorder_woff(const float* __restrict__ w_off,
                             unsigned short* __restrict__ wOffT) {
  int i = blockIdx.x * blockDim.x + threadIdx.x;
  if (i >= 32 * KDIM) return;
  int n = i / KDIM, k = i - n * KDIM;
  int tap = k >> 8, c = k & 255;
  float v = (n < 27) ? w_off[(n * C_ + c) * 9 + tap] : 0.0f;
  wOffT[i] = f2bfu(v);
}

__global__ void reorder_wdef(const float* __restrict__ w_def,
                             unsigned short* __restrict__ wBt) {
  int i = blockIdx.x * blockDim.x + threadIdx.x;
  if (i >= OC_ * KDIM) return;
  int oc = i / KDIM, k = i - oc * KDIM;
  int m = k >> 8, c = k & 255;
  wBt[i] = f2bfu(w_def[(oc * C_ + c) * 9 + m]);
}

// ---- stage 1: offset conv as bf16 WMMA GEMM (M=16384, N=32, K=2304) ------
__global__ __launch_bounds__(256) void offset_conv(
    const float* __restrict__ feat, const unsigned short* __restrict__ wOffT,
    const float* __restrict__ b_off,
    float* __restrict__ offY, float* __restrict__ offX, float* __restrict__ maskA)
{
  __shared__ __align__(16) unsigned short As[64][40];   // 64 pos x 32 k
  __shared__ __align__(16) unsigned short Bs[32][40];   // 32 n  x 32 k

  const int tid  = threadIdx.x;
  const int lane = tid & 31;
  const int wv   = tid >> 5;
  const int p0   = blockIdx.x * 64;
  const int b    = p0 >> 12;              // 4096 positions per batch image
  const float* fb0 = feat + (size_t)b * C_ * HW_;

  // A-build mapping: 32 lanes <-> 32 consecutive positions (coalesced)
  const int posl = ((wv & 1) << 5) | lane;     // 0..63
  const int cgrp = wv >> 1;                    // 0..3, 8 channels each
  const int rem  = (p0 + posl) & (HW_ - 1);
  const int py   = rem >> 6, px = rem & 63;

  const int sub_m = wv & 3;    // 4 M-subtiles (BM=64)
  const int sub_n = wv >> 2;   // 2 N-subtiles (BN=32)

  v8f acc = {};

  for (int tap = 0; tap < 9; ++tap) {
    const int dy = tap / 3 - 1, dx = tap % 3 - 1;
    const int yy = py + dy, xx = px + dx;
    // branch-free: clamped (always legal) address x {0,1} validity weight
    const float vms = ((yy >= 0) && (yy < H_) && (xx >= 0) && (xx < W_)) ? 1.0f : 0.0f;
    const int cy = min(max(yy, 0), H_ - 1), cx = min(max(xx, 0), W_ - 1);
    const int foff = cy * W_ + cx;
    for (int cc = 0; cc < 8; ++cc) {
      const int c0 = cc << 5;
      __syncthreads();   // previous iteration's WMMA reads done
      {
        unsigned short h[8];
#pragma unroll
        for (int j = 0; j < 8; ++j) {
          int c = c0 + cgrp * 8 + j;
          h[j] = f2bfu(fb0[(size_t)c * HW_ + foff] * vms);
        }
        uint4 pk;
        pk.x = (unsigned)h[0] | ((unsigned)h[1] << 16);
        pk.y = (unsigned)h[2] | ((unsigned)h[3] << 16);
        pk.z = (unsigned)h[4] | ((unsigned)h[5] << 16);
        pk.w = (unsigned)h[6] | ((unsigned)h[7] << 16);
        *reinterpret_cast<uint4*>(&As[posl][cgrp * 8]) = pk;
      }
      if (tid < 128) {
        int n = tid >> 2, seg = tid & 3;
        const uint4* src = reinterpret_cast<const uint4*>(
            wOffT + (size_t)n * KDIM + tap * 256 + c0 + seg * 8);
        *reinterpret_cast<uint4*>(&Bs[n][seg * 8]) = *src;
      }
      __syncthreads();
      v16bf a, bm;
      const int row = sub_m * 16 + (lane & 15);
      const int kb  = (lane >> 4) * 16;
#pragma unroll
      for (int j = 0; j < 8; ++j) {
        a[j]     = bfbits(As[row][kb + j]);
        a[j + 8] = bfbits(As[row][kb + 8 + j]);
      }
      const int nc  = sub_n * 16 + (lane & 15);
      const int kb2 = (lane >> 4) * 16;
#pragma unroll
      for (int j = 0; j < 16; ++j) bm[j] = bfbits(Bs[nc][kb2 + j]);
      acc = __builtin_amdgcn_wmma_f32_16x16x32_bf16(false, a, false, bm,
                                                    (short)0, acc, false, false);
    }
  }

  // epilogue: channels 0..17 interleaved off_y/off_x, 18..26 sigmoid mask
  const int nco = sub_n * 16 + (lane & 15);
  if (nco < 27) {
    const float bo = b_off[nco];
#pragma unroll
    for (int r = 0; r < 8; ++r) {
      int pl   = sub_m * 16 + ((lane >> 4) << 3) + r;
      int prem = (p0 + pl) & (HW_ - 1);
      float v  = acc[r] + bo;
      if (nco < 18) {
        int mi = nco >> 1;
        float* dst = (nco & 1) ? offX : offY;
        dst[(b * 9 + mi) * HW_ + prem] = v;
      } else {
        maskA[(b * 9 + (nco - 18)) * HW_ + prem] = 1.0f / (1.0f + __expf(-v));
      }
    }
  }
}

// ---- stage 2: fused bilinear gather + bf16 WMMA GEMM ---------------------
// Block: 32 positions x all 256 OC (gathers done exactly once per position).
// 8 waves: sub_m = wv&1 (2 tiles), each wave register-blocks 4 N-subtiles.
#define SMEM_AS    0                       // 32*40 ushort  = 2560 B
#define SMEM_BS    2560                    // 256*40 ushort = 20480 B
#define SMEM_SO    (2560 + 20480)          // 32*4 int      = 512 B
#define SMEM_SW    (2560 + 20480 + 512)    // 32*4 float    = 512 B
#define SMEM_POOL  33792                   // epilogue Cs: 256*33*4 = 33792 B

__global__ __launch_bounds__(256) void dcn_main(
    const float* __restrict__ feat,
    const unsigned short* __restrict__ wBt,
    const float* __restrict__ offY, const float* __restrict__ offX,
    const float* __restrict__ maskA, const float* __restrict__ b_def,
    float* __restrict__ out)
{
  __shared__ __align__(16) unsigned char smem[SMEM_POOL];
  unsigned short* As  = (unsigned short*)(smem + SMEM_AS);   // [32][40]
  unsigned short* Bs  = (unsigned short*)(smem + SMEM_BS);   // [256][40]
  int*            sO  = (int*)(smem + SMEM_SO);              // [32][4]
  float*          sWt = (float*)(smem + SMEM_SW);            // [32][4]
  float*          Cs  = (float*)smem;                        // [256][33] (epilogue)

  const int tid   = threadIdx.x;
  const int lane  = tid & 31;
  const int wv    = tid >> 5;
  const int p0    = blockIdx.x * 32;
  const int b     = p0 >> 12;
  const int prem0 = p0 & (HW_ - 1);
  const float* fb0 = feat + (size_t)b * C_ * HW_;

  const int sub_m  = wv & 1;         // 2 M-subtiles (BM=32)
  const int nbase  = (wv >> 1) * 64; // each wave: N-subtiles nbase + {0,16,32,48}

  v8f acc0 = {}, acc1 = {}, acc2 = {}, acc3 = {};

  for (int m = 0; m < 9; ++m) {
    __syncthreads();   // all gathers/WMMAs of previous m done
    if (tid < 32) {
      int prem = prem0 + tid;
      int py = prem >> 6, px = prem & 63;
      int base = (b * 9 + m) * HW_ + prem;
      float oy = offY[base], ox = offX[base], mk = maskA[base];
      float sy = oy + (float)(py + m / 3 - 1);
      float sx = ox + (float)(px + m % 3 - 1);
      float fy = floorf(sy), fx = floorf(sx);
      int y0 = (int)fy, x0 = (int)fx;
      float wy1 = sy - fy, wy0 = 1.0f - wy1;
      float wx1 = sx - fx, wx0 = 1.0f - wx1;
      int y1 = y0 + 1, x1 = x0 + 1;
      bool vy0 = (y0 >= 0) && (y0 < H_), vy1 = (y1 >= 0) && (y1 < H_);
      bool vx0 = (x0 >= 0) && (x0 < W_), vx1 = (x1 >= 0) && (x1 < W_);
      int cy0 = min(max(y0, 0), H_ - 1), cy1 = min(max(y1, 0), H_ - 1);
      int cx0 = min(max(x0, 0), W_ - 1), cx1 = min(max(x1, 0), W_ - 1);
      sO[tid * 4 + 0] = cy0 * W_ + cx0;  sO[tid * 4 + 1] = cy0 * W_ + cx1;
      sO[tid * 4 + 2] = cy1 * W_ + cx0;  sO[tid * 4 + 3] = cy1 * W_ + cx1;
      sWt[tid * 4 + 0] = (vy0 && vx0) ? mk * wy0 * wx0 : 0.0f;
      sWt[tid * 4 + 1] = (vy0 && vx1) ? mk * wy0 * wx1 : 0.0f;
      sWt[tid * 4 + 2] = (vy1 && vx0) ? mk * wy1 * wx0 : 0.0f;
      sWt[tid * 4 + 3] = (vy1 && vx1) ? mk * wy1 * wx1 : 0.0f;
    }
    __syncthreads();
    // lane <-> position: corner addresses of adjacent lanes are adjacent
    const int o0 = sO[lane * 4 + 0], o1 = sO[lane * 4 + 1];
    const int o2 = sO[lane * 4 + 2], o3 = sO[lane * 4 + 3];
    const float g0 = sWt[lane * 4 + 0], g1 = sWt[lane * 4 + 1];
    const float g2 = sWt[lane * 4 + 2], g3 = sWt[lane * 4 + 3];

    for (int cc = 0; cc < 8; ++cc) {
      const int c0 = cc << 5;
      // build A: each wave gathers 4 channels for all 32 positions
      {
        unsigned short h[4];
#pragma unroll
        for (int j = 0; j < 4; ++j) {
          int c = c0 + wv * 4 + j;
          const float* fc = fb0 + (size_t)c * HW_;
          h[j] = f2bfu(g0 * fc[o0] + g1 * fc[o1] + g2 * fc[o2] + g3 * fc[o3]);
        }
        uint2 pk;
        pk.x = (unsigned)h[0] | ((unsigned)h[1] << 16);
        pk.y = (unsigned)h[2] | ((unsigned)h[3] << 16);
        *reinterpret_cast<uint2*>(&As[lane * 40 + wv * 4]) = pk;
      }
      // load B tile: one full 64B row of 32 bf16 per thread (16KB total)
      {
        const uint4* src = reinterpret_cast<const uint4*>(
            wBt + (size_t)tid * KDIM + m * 256 + c0);
        uint4* dst = reinterpret_cast<uint4*>(&Bs[tid * 40]);
#pragma unroll
        for (int q = 0; q < 4; ++q) dst[q] = src[q];
      }
      __syncthreads();
      v16bf a;
      const int row = sub_m * 16 + (lane & 15);
      const int kb  = (lane >> 4) * 16;
#pragma unroll
      for (int j = 0; j < 8; ++j) {
        a[j]     = bfbits(As[row * 40 + kb + j]);
        a[j + 8] = bfbits(As[row * 40 + kb + 8 + j]);
      }
      const int kb2 = (lane >> 4) * 16;
      const int nr  = nbase + (lane & 15);
#pragma unroll
      for (int q = 0; q < 4; ++q) {
        v16bf bm;
#pragma unroll
        for (int j = 0; j < 16; ++j) bm[j] = bfbits(Bs[(nr + q * 16) * 40 + kb2 + j]);
        if (q == 0) acc0 = __builtin_amdgcn_wmma_f32_16x16x32_bf16(false, a, false, bm, (short)0, acc0, false, false);
        if (q == 1) acc1 = __builtin_amdgcn_wmma_f32_16x16x32_bf16(false, a, false, bm, (short)0, acc1, false, false);
        if (q == 2) acc2 = __builtin_amdgcn_wmma_f32_16x16x32_bf16(false, a, false, bm, (short)0, acc2, false, false);
        if (q == 3) acc3 = __builtin_amdgcn_wmma_f32_16x16x32_bf16(false, a, false, bm, (short)0, acc3, false, false);
      }
      __syncthreads();   // guard A/B overwrite by next chunk
    }
  }

  // epilogue: stage through LDS (reusing A/B pool), coalesced 128B stores
  __syncthreads();
  {
    const int nc0 = nbase + (lane & 15);
    const int pb  = sub_m * 16 + ((lane >> 4) << 3);
#pragma unroll
    for (int r = 0; r < 8; ++r) {
      Cs[(nc0 +  0) * 33 + pb + r] = acc0[r];
      Cs[(nc0 + 16) * 33 + pb + r] = acc1[r];
      Cs[(nc0 + 32) * 33 + pb + r] = acc2[r];
      Cs[(nc0 + 48) * 33 + pb + r] = acc3[r];
    }
  }
  __syncthreads();
  {
    const float bd = b_def[tid];
    float* dst = out + (size_t)(b * OC_ + tid) * HW_ + prem0;
#pragma unroll
    for (int q = 0; q < 8; ++q) {
      float4 v;
      v.x = Cs[tid * 33 + q * 4 + 0] + bd;
      v.y = Cs[tid * 33 + q * 4 + 1] + bd;
      v.z = Cs[tid * 33 + q * 4 + 2] + bd;
      v.w = Cs[tid * 33 + q * 4 + 3] + bd;
      *reinterpret_cast<float4*>(dst + q * 4) = v;
    }
  }
}

// ---- host launch ---------------------------------------------------------
extern "C" void kernel_launch(void* const* d_in, const int* in_sizes, int n_in,
                              void* d_out, int out_size, void* d_ws, size_t ws_size,
                              hipStream_t stream) {
  const float* feat  = (const float*)d_in[0];
  const float* w_off = (const float*)d_in[1];
  const float* b_off = (const float*)d_in[2];
  const float* w_def = (const float*)d_in[3];
  const float* b_def = (const float*)d_in[4];
  float* out = (float*)d_out;

  // workspace layout (≈3.0 MB total)
  float* offY  = (float*)d_ws;                       // 4*9*4096
  float* offX  = offY + B_ * 9 * HW_;
  float* maskA = offX + B_ * 9 * HW_;
  unsigned short* wOffT = (unsigned short*)(maskA + B_ * 9 * HW_);  // 32*2304 bf16
  unsigned short* wBt   = wOffT + 32 * KDIM;                        // 256*2304 bf16

  reorder_woff<<<(32 * KDIM + 255) / 256, 256, 0, stream>>>(w_off, wOffT);
  reorder_wdef<<<(OC_ * KDIM + 255) / 256, 256, 0, stream>>>(w_def, wBt);
  offset_conv<<<(B_ * HW_) / 64, 256, 0, stream>>>(feat, wOffT, b_off,
                                                   offY, offX, maskA);
  dcn_main<<<(B_ * HW_) / 32, 256, 0, stream>>>(
      feat, wBt, offY, offX, maskA, b_def, out);
}
